// Model_22110491640669
// MI455X (gfx1250) — matrix operand order repaired and checked by reading, not compiled, not run on hardware
//
#include <hip/hip_runtime.h>

// ---------------- model dims ----------------
#define NB 32            // batches (groups)
#define NG 128           // group size
#define NA 32            // attended rows per group
#define NN 4096          // NB*NG rows
#define LL 64            // tokens per row
#define DE 768           // embedding dim
#define DD 256           // model dim
#define NH 8             // heads
#define DH 32            // head dim
#define NE 65536         // edges per graph
#define HIDN 512
#define EPSC 0.3f

typedef __bf16 bf16;
typedef bf16 v16bf __attribute__((ext_vector_type(16)));
typedef bf16 v8bf  __attribute__((ext_vector_type(8)));
typedef float v8f  __attribute__((ext_vector_type(8)));
typedef unsigned int u32x4 __attribute__((ext_vector_type(4)));
typedef int i32x8 __attribute__((ext_vector_type(8)));
typedef int i32x4 __attribute__((ext_vector_type(4)));

// =====================================================================
// Register-direct GEMM: C[M,N] = act(A[M,K](bf16) * W[N,K]^T(bf16) + bias)
// One wave per 16x16 tile, K-loop of 32 with v_wmma_f32_16x16x32_bf16.
// =====================================================================
__global__ void gemm_bf16_wmma(const bf16* __restrict__ A, const bf16* __restrict__ Bw,
                               const float* __restrict__ bias, float* __restrict__ C,
                               int M, int N, int K, int act)
{
    int wave = (int)((blockIdx.x * blockDim.x + threadIdx.x) >> 5);
    int lane = (int)(threadIdx.x & 31);
    int tilesN = N >> 4;
    int tiles  = (M >> 4) * tilesN;
    if (wave >= tiles) return;                 // wave-uniform: EXEC stays all-ones
    int mt = wave / tilesN, nt = wave % tilesN;
    int m0 = mt << 4, n0 = nt << 4;
    int kh = lane >> 4;                        // K-half selector
    int ml = lane & 15;                        // row (A) / col (B) index

    const bf16* arow = A  + (size_t)(m0 + ml) * K + kh * 8;
    const bf16* brow = Bw + (size_t)(n0 + ml) * K + kh * 8;

    v8f acc = {};
    for (int k0 = 0; k0 < K; k0 += 32) {
        __builtin_prefetch(arow + k0 + 128, 0, 3);   // global_prefetch_b8, near-cache
        __builtin_prefetch(brow + k0 + 128, 0, 3);
        v8bf alo = *(const v8bf*)(arow + k0);
        v8bf ahi = *(const v8bf*)(arow + k0 + 16);
        v16bf a;
        #pragma unroll
        for (int i = 0; i < 8; ++i) { a[i] = alo[i]; a[i + 8] = ahi[i]; }
        v8bf blo = *(const v8bf*)(brow + k0);
        v8bf bhi = *(const v8bf*)(brow + k0 + 16);
        v16bf b;
        #pragma unroll
        for (int i = 0; i < 8; ++i) { b[i] = blo[i]; b[i + 8] = bhi[i]; }
        acc = __builtin_amdgcn_wmma_f32_16x16x32_bf16(false, a, false, b,
                                                      (short)0, acc, false, false);
    }
    float bv = bias ? bias[n0 + ml] : 0.0f;
    int rbase = kh * 8;
    #pragma unroll
    for (int i = 0; i < 8; ++i) {
        float v = acc[i] + bv;
        if (act == 1)      v = v > 0.0f ? v : 0.0f;
        else if (act == 2) v = tanhf(v);
        C[(size_t)(m0 + rbase + i) * N + (n0 + ml)] = v;
    }
}

// =====================================================================
// TDM-staged GEMM: block tile 64x64, K-chunks of 64 staged into LDS by the
// Tensor Data Mover (tensor_load_to_lds + s_wait_tensorcnt), with TDM LDS
// padding (8 DWORDs per 128B row) -> LDS row stride 160B (keeps ds_load
// 16B/32B alignment, de-periodizes banks). 8 waves, each 16x32 sub-tile.
// =====================================================================
__device__ __forceinline__ void tdm_issue_2d(unsigned long long ga, unsigned int lds,
                                             int rows, int K)
{
    u32x4 g0v;
    g0v[0] = 1u;                                   // count=1, user descriptor
    g0v[1] = lds;                                  // LDS byte address
    g0v[2] = (unsigned int)ga;                     // global_addr[31:0]
    g0v[3] = (unsigned int)(ga >> 32) | (2u << 30);// global_addr[56:32] | type=2
    i32x8 g1v;
    // data_size=2B(code1), pad_enable, pad_interval=128B(code4), pad_amount=8dw(code7)
    g1v[0] = (int)((1u << 16) | (1u << 20) | (4u << 22) | (7u << 25));
    g1v[1] = (int)(((unsigned int)K & 0xFFFFu) << 16);               // tensor_dim0 lo
    g1v[2] = (int)(((unsigned int)K >> 16) |
                   (((unsigned int)rows & 0xFFFFu) << 16));          // dim0 hi | dim1 lo
    g1v[3] = (int)(((unsigned int)rows >> 16) | (64u << 16));        // dim1 hi | tile_dim0=64
    g1v[4] = 64;                                   // tile_dim1=64, tile_dim2=0
    g1v[5] = K;                                    // tensor_dim0_stride lo
    g1v[6] = 0;
    g1v[7] = 0;
    i32x4 z4 = {0, 0, 0, 0};
#if __clang_major__ >= 23
    i32x8 z8 = {0, 0, 0, 0, 0, 0, 0, 0};
    __builtin_amdgcn_tensor_load_to_lds(g0v, g1v, z4, z4, z8, 0);
#else
    __builtin_amdgcn_tensor_load_to_lds(g0v, g1v, z4, z4, 0);
#endif
}

#define LDSTRIDE 80   // 64 elems + 8-DWORD TDM pad = 160 bytes

__global__ void gemm_tdm_wmma(const bf16* __restrict__ A, const bf16* __restrict__ Bw,
                              const float* __restrict__ bias, float* __restrict__ C,
                              int M, int N, int K, int act)
{
    __shared__ __align__(64) bf16 Ab[64 * LDSTRIDE];
    __shared__ __align__(64) bf16 Bb[64 * LDSTRIDE];
    int nb = N >> 6;
    int bm = (int)blockIdx.x / nb, bn = (int)blockIdx.x % nb;
    int m0 = bm << 6, n0 = bn << 6;
    int wv = (int)(threadIdx.x >> 5);
    int lane = (int)(threadIdx.x & 31);
    int mw = wv >> 1, nw = wv & 1;                 // 4x2 wave grid over 64x64 tile
    int kh = lane >> 4, ml = lane & 15;

    unsigned int ldsA = (unsigned int)(size_t)&Ab[0];   // flat addr low 32 = LDS offset
    unsigned int ldsB = (unsigned int)(size_t)&Bb[0];

    v8f acc0 = {}, acc1 = {};
    for (int k0 = 0; k0 < K; k0 += 64) {
        if (wv == 0) {
            unsigned long long gaA = (unsigned long long)(size_t)A
                                   + 2ull * ((unsigned long long)m0 * K + k0);
            unsigned long long gaB = (unsigned long long)(size_t)Bw
                                   + 2ull * ((unsigned long long)n0 * K + k0);
            tdm_issue_2d(gaA, ldsA, M, K);
            tdm_issue_2d(gaB, ldsB, N, K);
            __builtin_amdgcn_s_wait_tensorcnt(0);
        }
        __syncthreads();
        #pragma unroll
        for (int kk = 0; kk < 64; kk += 32) {
            const bf16* ap = &Ab[(mw * 16 + ml) * LDSTRIDE + kk + kh * 8];
            v8bf alo = *(const v8bf*)ap;
            v8bf ahi = *(const v8bf*)(ap + 16);
            v16bf a;
            #pragma unroll
            for (int i = 0; i < 8; ++i) { a[i] = alo[i]; a[i + 8] = ahi[i]; }
            const bf16* bp0 = &Bb[(nw * 32 + ml)      * LDSTRIDE + kk + kh * 8];
            const bf16* bp1 = &Bb[(nw * 32 + 16 + ml) * LDSTRIDE + kk + kh * 8];
            v8bf b0lo = *(const v8bf*)bp0, b0hi = *(const v8bf*)(bp0 + 16);
            v8bf b1lo = *(const v8bf*)bp1, b1hi = *(const v8bf*)(bp1 + 16);
            v16bf b0, b1;
            #pragma unroll
            for (int i = 0; i < 8; ++i) { b0[i] = b0lo[i]; b0[i + 8] = b0hi[i]; b1[i] = b1lo[i]; b1[i + 8] = b1hi[i]; }
            acc0 = __builtin_amdgcn_wmma_f32_16x16x32_bf16(false, a, false, b0,
                                                           (short)0, acc0, false, false);
            acc1 = __builtin_amdgcn_wmma_f32_16x16x32_bf16(false, a, false, b1,
                                                           (short)0, acc1, false, false);
        }
        __syncthreads();
    }
    int row0 = m0 + mw * 16 + kh * 8;
    int col0 = n0 + nw * 32 + ml;
    float bv0 = bias[col0], bv1 = bias[col0 + 16];
    #pragma unroll
    for (int i = 0; i < 8; ++i) {
        float v0 = acc0[i] + bv0, v1 = acc1[i] + bv1;
        if (act == 1)      { v0 = fmaxf(v0, 0.f); v1 = fmaxf(v1, 0.f); }
        else if (act == 2) { v0 = tanhf(v0);      v1 = tanhf(v1); }
        C[(size_t)(row0 + i) * N + col0]      = v0;
        C[(size_t)(row0 + i) * N + col0 + 16] = v1;
    }
}

// ---------------- small helpers ----------------
__global__ void f32_to_bf16_k(const float* __restrict__ in, bf16* __restrict__ out, int n) {
    int i = blockIdx.x * blockDim.x + threadIdx.x;
    if (i < n) out[i] = (bf16)in[i];
}
__global__ void zero_f32_k(float* p, int n) {
    int i = blockIdx.x * blockDim.x + threadIdx.x;
    if (i < n) p[i] = 0.0f;
}
__global__ void copy_f32_k(const float* __restrict__ in, float* __restrict__ out, int n) {
    int i = blockIdx.x * blockDim.x + threadIdx.x;
    if (i < n) out[i] = in[i];
}

__device__ __forceinline__ float block_reduce_256(float v, float* red) {
    int tid = threadIdx.x;
    red[tid] = v; __syncthreads();
    for (int s = 128; s > 0; s >>= 1) {
        if (tid < s) red[tid] += red[tid + s];
        __syncthreads();
    }
    float r = red[0]; __syncthreads();
    return r;
}

// ---------------- stage 1: embedding masked-mean pool ----------------
__global__ void embed_pool_k(const int* __restrict__ ids, const float* __restrict__ mask,
                             const float* __restrict__ emb, bf16* __restrict__ pooled_bf)
{
    int n = blockIdx.x, tid = threadIdx.x;
    float a0 = 0.f, a1 = 0.f, a2 = 0.f, ms = 0.f;
    for (int t = 0; t < LL; ++t) {
        float mv = mask[(size_t)n * LL + t];
        int id = ids[(size_t)n * LL + t];
        const float* e = emb + (size_t)id * DE;
        a0 += e[tid] * mv; a1 += e[tid + 256] * mv; a2 += e[tid + 512] * mv;
        ms += mv;
    }
    float inv = 1.0f / fmaxf(ms, 1.0f);
    size_t base = (size_t)n * DE;
    pooled_bf[base + tid]       = (bf16)(a0 * inv);
    pooled_bf[base + tid + 256] = (bf16)(a1 * inv);
    pooled_bf[base + tid + 512] = (bf16)(a2 * inv);
}

// ---------------- AHT: gather attended query rows ----------------
__global__ void gather_q_k(const int* __restrict__ add_ids, const float* __restrict__ x,
                           bf16* __restrict__ qin)
{
    int ba = blockIdx.x;              // 0..NB*NA-1
    int b = ba >> 5, a = ba & 31;
    int row = add_ids[b * NA + a];
    qin[(size_t)ba * DD + threadIdx.x] = (bf16)x[((size_t)(b * NG + row)) * DD + threadIdx.x];
}

// ---------------- MAB attention: per (group, head), 128 threads --------
__global__ void mab_attn_k(const float* __restrict__ Q, const float* __restrict__ Kall,
                           const float* __restrict__ Vall, float* __restrict__ out)
{
    int b = blockIdx.x >> 3, h = blockIdx.x & 7;
    int tid = threadIdx.x;            // 0..127
    __shared__ float sc[NA * NG];     // 16 KB
    for (int idx = tid; idx < NA * NG; idx += 128) {
        int ai = idx >> 7, g = idx & 127;
        const float* qp = Q    + ((size_t)(b * NA + ai)) * DD + h * DH;
        const float* kp = Kall + ((size_t)(b * NG + g )) * DD + h * DH;
        float s = 0.f;
        #pragma unroll 8
        for (int k = 0; k < DH; ++k) s += qp[k] * kp[k];
        sc[idx] = s * (1.0f / 16.0f);            // 1/sqrt(D=256)
    }
    __syncthreads();
    if (tid < NA) {
        float mx = -1e30f;
        for (int g = 0; g < NG; ++g) mx = fmaxf(mx, sc[tid * NG + g]);
        float sum = 0.f;
        for (int g = 0; g < NG; ++g) { float e = expf(sc[tid * NG + g] - mx); sc[tid * NG + g] = e; sum += e; }
        float inv = 1.0f / sum;
        for (int g = 0; g < NG; ++g) sc[tid * NG + g] *= inv;
    }
    __syncthreads();
    for (int idx = tid; idx < NA * DH; idx += 128) {
        int ai = idx >> 5, d = idx & 31;
        float o = Q[((size_t)(b * NA + ai)) * DD + h * DH + d];
        for (int g = 0; g < NG; ++g)
            o += sc[ai * NG + g] * Vall[((size_t)(b * NG + g)) * DD + h * DH + d];
        out[((size_t)(b * NA + ai)) * DD + h * DH + d] = o;
    }
}

// ---------------- LayerNorm over dim 256 (optionally fused add) --------
__global__ void ln_k(const float* __restrict__ in, const float* __restrict__ addv,
                     const float* __restrict__ gamma, const float* __restrict__ beta,
                     float* __restrict__ outf, bf16* __restrict__ outb)
{
    int row = blockIdx.x, tid = threadIdx.x;
    __shared__ float red[256];
    float v = in[(size_t)row * DD + tid];
    if (addv) v += addv[(size_t)row * DD + tid];
    float mean = block_reduce_256(v, red) * (1.0f / 256.0f);
    float dv = v - mean;
    float var = block_reduce_256(dv * dv, red) * (1.0f / 256.0f);
    float y = dv * rsqrtf(var + 1e-5f) * gamma[tid] + beta[tid];
    outf[(size_t)row * DD + tid] = y;
    if (outb) outb[(size_t)row * DD + tid] = (bf16)y;
}

// ---------------- AHT scatter: x2[b, aid] += xh ----------------
__global__ void scatter_add_k(const int* __restrict__ add_ids, const float* __restrict__ xh,
                              float* __restrict__ x2)
{
    int ba = blockIdx.x;
    int b = ba >> 5, a = ba & 31;
    int row = add_ids[b * NA + a];
    x2[((size_t)(b * NG + row)) * DD + threadIdx.x] += xh[(size_t)ba * DD + threadIdx.x];
}

// ---------------- FAGCN pieces ----------------
__global__ void deg_k(const int* __restrict__ dst, float* __restrict__ deg) {
    int e = blockIdx.x * blockDim.x + threadIdx.x;
    if (e < NE) atomicAdd(&deg[dst[e]], 1.0f);
}
__global__ void dinv_k(const float* __restrict__ deg, float* __restrict__ dinv) {
    int i = blockIdx.x * blockDim.x + threadIdx.x;
    if (i < NN) { float d = deg[i]; dinv[i] = d > 0.f ? rsqrtf(fmaxf(d, 1.0f)) : 0.0f; }
}
__global__ void alar_k(const float* __restrict__ x, const float* __restrict__ Wal,
                       const float* __restrict__ balp, const float* __restrict__ War,
                       const float* __restrict__ barp, float* __restrict__ al,
                       float* __restrict__ ar)
{
    int wave = (int)((blockIdx.x * blockDim.x + threadIdx.x) >> 5);
    int lane = (int)(threadIdx.x & 31);
    if (wave >= NN) return;
    const float* xr = x + (size_t)wave * DD;
    float s1 = 0.f, s2 = 0.f;
    for (int k = lane; k < DD; k += 32) { float xv = xr[k]; s1 += xv * Wal[k]; s2 += xv * War[k]; }
    for (int m = 16; m >= 1; m >>= 1) { s1 += __shfl_xor(s1, m, 32); s2 += __shfl_xor(s2, m, 32); }
    if (lane == 0) { al[wave] = s1 + balp[0]; ar[wave] = s2 + barp[0]; }
}
__global__ void edge_k(const int* __restrict__ src, const int* __restrict__ dst,
                       const float* __restrict__ al, const float* __restrict__ ar,
                       const float* __restrict__ dinv, const float* __restrict__ x,
                       float* __restrict__ agg)
{
    int e = blockIdx.x;
    int s = src[e], t = dst[e];
    float w = tanhf(al[s] + ar[t]) * dinv[s] * dinv[t];
    atomicAdd(&agg[(size_t)t * DD + threadIdx.x], x[(size_t)s * DD + threadIdx.x] * w);
}
__global__ void combine_k(const float* __restrict__ raw, const float* __restrict__ agg,
                          float* __restrict__ out, int n)
{
    int i = blockIdx.x * blockDim.x + threadIdx.x;
    if (i < n) { float v = EPSC * raw[i] + agg[i]; out[i] = v > 0.f ? v : 0.f; }
}

// ---------------- gating softmax + expert mix + concat (bf16 out) ------
__global__ void gating_concat_k(const float* __restrict__ x2, const float* __restrict__ Wga,
                                const float* __restrict__ bga, const float* __restrict__ Wgb,
                                const float* __restrict__ bgb, const float* __restrict__ fa,
                                const float* __restrict__ fs, const float* __restrict__ fb,
                                bf16* __restrict__ gcat)
{
    int n = blockIdx.x, tid = threadIdx.x;
    __shared__ float red[256];
    __shared__ float gg[4];
    float xv = x2[(size_t)n * DD + tid];
    float u0 = block_reduce_256(xv * Wga[tid],       red);
    float u1 = block_reduce_256(xv * Wga[256 + tid], red);
    float w0 = block_reduce_256(xv * Wgb[tid],       red);
    float w1 = block_reduce_256(xv * Wgb[256 + tid], red);
    if (tid == 0) {
        u0 += bga[0]; u1 += bga[1]; w0 += bgb[0]; w1 += bgb[1];
        float m = fmaxf(u0, u1), e0 = expf(u0 - m), e1 = expf(u1 - m);
        gg[0] = e0 / (e0 + e1); gg[1] = e1 / (e0 + e1);
        m = fmaxf(w0, w1); e0 = expf(w0 - m); e1 = expf(w1 - m);
        gg[2] = e0 / (e0 + e1); gg[3] = e1 / (e0 + e1);
    }
    __syncthreads();
    size_t r = (size_t)n * DD;
    gcat[(size_t)n * 512 + tid]       = (bf16)(gg[0] * fa[r + tid] + gg[1] * fs[r + tid]);
    gcat[(size_t)n * 512 + 256 + tid] = (bf16)(gg[2] * fb[r + tid] + gg[3] * fs[r + tid]);
}

// ---------------- fuse: per-group MHA (query row 0 only) ---------------
__global__ void fuse_k(const float* __restrict__ qkv, const float* __restrict__ Wout,
                       const float* __restrict__ bout, float* __restrict__ vec,
                       float* __restrict__ scores)
{
    int b = blockIdx.x, tid = threadIdx.x;     // 128 threads
    __shared__ float att[NG];
    __shared__ float red[NG];
    __shared__ float o0[DD];
    const float* q0 = qkv + (size_t)(b * NG) * (3 * DD);  // row 0, cols [0,256)
    float ssum = 0.f;
    for (int h = 0; h < NH; ++h) {
        const float* kg = qkv + (size_t)(b * NG + tid) * (3 * DD) + DD + h * DH;
        const float* qh = q0 + h * DH;
        float s = 0.f;
        #pragma unroll 8
        for (int k = 0; k < DH; ++k) s += qh[k] * kg[k];
        s *= 0.17677669529663689f;             // 1/sqrt(DH=32)
        red[tid] = s; __syncthreads();
        for (int st = 64; st > 0; st >>= 1) { if (tid < st) red[tid] = fmaxf(red[tid], red[tid + st]); __syncthreads(); }
        float mx = red[0]; __syncthreads();
        float e = expf(s - mx);
        red[tid] = e; __syncthreads();
        for (int st = 64; st > 0; st >>= 1) { if (tid < st) red[tid] += red[tid + st]; __syncthreads(); }
        float aval = e / red[0];
        att[tid] = aval; ssum += aval;
        __syncthreads();
        if (tid < DH) {
            float o = 0.f;
            for (int g = 0; g < NG; ++g)
                o += att[g] * qkv[(size_t)(b * NG + g) * (3 * DD) + 2 * DD + h * DH + tid];
            o0[h * DH + tid] = o;
        }
        __syncthreads();
    }
    scores[(size_t)b * NG + tid] = ssum * (1.0f / NH);
    for (int d = tid; d < DD; d += 128) {
        float v = bout[d];
        for (int k = 0; k < DD; ++k) v += o0[k] * Wout[(size_t)d * DD + k];
        vec[(size_t)b * DD + d] = v;
    }
}

// ---------------- final MLP head ----------------
__global__ void head_k(const float* __restrict__ vec, const float* __restrict__ W1,
                       const float* __restrict__ b1, const float* __restrict__ W2,
                       const float* __restrict__ b2, float* __restrict__ pred)
{
    int b = blockIdx.x, tid = threadIdx.x;    // 256 threads
    __shared__ float red[256];
    const float* v = vec + (size_t)b * DD;
    float part = 0.f;
    for (int r = 0; r < 2; ++r) {
        int j = tid + r * 256;
        float h = b1[j];
        for (int k = 0; k < DD; ++k) h += v[k] * W1[(size_t)j * DD + k];
        h = fmaxf(h, 0.f);
        part += h * W2[j];
    }
    float tot = block_reduce_256(part, red);
    if (tid == 0) pred[b] = 1.0f / (1.0f + expf(-(tot + b2[0])));
}

// =====================================================================
extern "C" void kernel_launch(void* const* d_in, const int* in_sizes, int n_in,
                              void* d_out, int out_size, void* d_ws, size_t ws_size,
                              hipStream_t stream)
{
    (void)in_sizes; (void)n_in; (void)out_size; (void)ws_size;
    const int*   input_ids   = (const int*)d_in[0];
    const float* input_masks = (const float*)d_in[1];
    const int*   e_g0 = (const int*)d_in[2];
    const int*   e_g1 = (const int*)d_in[3];
    const int*   e_g2 = (const int*)d_in[4];
    const int*   add_ids = (const int*)d_in[5];
    const float* tok_emb = (const float*)d_in[6];
    const float *Wpool=(const float*)d_in[7],  *bpool=(const float*)d_in[8];
    const float *Wproj=(const float*)d_in[9],  *bproj=(const float*)d_in[10];
    const float *Wq=(const float*)d_in[11],    *bq=(const float*)d_in[12];
    const float *Wk=(const float*)d_in[13],    *bk=(const float*)d_in[14];
    const float *Wv=(const float*)d_in[15],    *bv=(const float*)d_in[16];
    const float *ln_g1=(const float*)d_in[17], *ln_b1=(const float*)d_in[18];
    const float *ln_g2=(const float*)d_in[19], *ln_b2=(const float*)d_in[20];
    const float *Wmab=(const float*)d_in[21],  *bmab=(const float*)d_in[22];
    const float *Wal0=(const float*)d_in[23],  *bal0=(const float*)d_in[24];
    const float *War0=(const float*)d_in[25],  *bar0=(const float*)d_in[26];
    const float *Wal1=(const float*)d_in[27],  *bal1=(const float*)d_in[28];
    const float *War1=(const float*)d_in[29],  *bar1=(const float*)d_in[30];
    const float *Wga=(const float*)d_in[31],   *bga=(const float*)d_in[32];
    const float *Wgb=(const float*)d_in[33],   *bgb=(const float*)d_in[34];
    const float *Wfl=(const float*)d_in[35],   *bfl=(const float*)d_in[36];
    const float *Win=(const float*)d_in[37],   *bin=(const float*)d_in[38];
    const float *Wout=(const float*)d_in[39],  *bout=(const float*)d_in[40];
    const float *W1=(const float*)d_in[41],    *b1=(const float*)d_in[42];
    const float *W2=(const float*)d_in[43],    *b2=(const float*)d_in[44];

    float* pred   = (float*)d_out;
    float* scores = (float*)d_out + NB;

    // ---------------- workspace bump allocator ----------------
    uint8_t* wsb = (uint8_t*)d_ws; size_t off = 0;
    auto alloc = [&](size_t bytes) -> void* {
        void* p = wsb + off;
        off += (bytes + 255) & ~(size_t)255;
        return p;
    };
    bf16*  pooled_bf = (bf16*) alloc((size_t)NN * DE * 2);
    float* h1        = (float*)alloc((size_t)NN * DE * 4);
    bf16*  h1_bf     = (bf16*) alloc((size_t)NN * DE * 2);
    float* xf        = (float*)alloc((size_t)NN * DD * 4);
    bf16*  x_bf      = (bf16*) alloc((size_t)NN * DD * 2);
    bf16*  Wpool_bf  = (bf16*) alloc((size_t)DE * DE * 2);
    bf16*  Wproj_bf  = (bf16*) alloc((size_t)DD * DE * 2);
    bf16*  Wq_bf     = (bf16*) alloc((size_t)DD * DD * 2);
    bf16*  Wk_bf     = (bf16*) alloc((size_t)DD * DD * 2);
    bf16*  Wv_bf     = (bf16*) alloc((size_t)DD * DD * 2);
    bf16*  Wmab_bf   = (bf16*) alloc((size_t)DD * DD * 2);
    bf16*  Wfl_bf    = (bf16*) alloc((size_t)DD * 512 * 2);
    bf16*  Win_bf    = (bf16*) alloc((size_t)(3 * DD) * DD * 2);
    float* Kall      = (float*)alloc((size_t)NN * DD * 4);
    float* Vall      = (float*)alloc((size_t)NN * DD * 4);
    bf16*  qin_bf    = (bf16*) alloc((size_t)NB * NA * DD * 2);
    float* Qf        = (float*)alloc((size_t)NB * NA * DD * 4);
    float* attn_o    = (float*)alloc((size_t)NB * NA * DD * 4);
    float* oln       = (float*)alloc((size_t)NB * NA * DD * 4);
    bf16*  oln_bf    = (bf16*) alloc((size_t)NB * NA * DD * 2);
    float* ffb       = (float*)alloc((size_t)NB * NA * DD * 4);
    float* xh        = (float*)alloc((size_t)NB * NA * DD * 4);
    float* x2        = (float*)alloc((size_t)NN * DD * 4);
    float* deg       = (float*)alloc((size_t)NN * 4);
    float* dinv      = (float*)alloc((size_t)NN * 4);
    float* al        = (float*)alloc((size_t)NN * 4);
    float* ar        = (float*)alloc((size_t)NN * 4);
    float* agg       = (float*)alloc((size_t)NN * DD * 4);
    float* fs        = (float*)alloc((size_t)NN * DD * 4);
    float* fa        = (float*)alloc((size_t)NN * DD * 4);
    float* fb        = (float*)alloc((size_t)NN * DD * 4);
    bf16*  gcat_bf   = (bf16*) alloc((size_t)NN * 512 * 2);
    float* xx        = (float*)alloc((size_t)NN * DD * 4);
    bf16*  xx_bf     = (bf16*) alloc((size_t)NN * DD * 2);
    float* qkv       = (float*)alloc((size_t)NN * 3 * DD * 4);
    float* vec       = (float*)alloc((size_t)NB * DD * 4);

    auto gemm = [&](const bf16* A, const bf16* Bw, const float* bias, float* C,
                    int M, int N, int K, int act) {
        if ((M % 64 == 0) && (N % 64 == 0) && (K % 64 == 0) && M >= 2048) {
            int blocks = (M / 64) * (N / 64);        // TDM-staged 64x64 block tiles
            gemm_tdm_wmma<<<blocks, 256, 0, stream>>>(A, Bw, bias, C, M, N, K, act);
        } else {
            int tiles = (M / 16) * (N / 16);
            gemm_bf16_wmma<<<(tiles + 7) / 8, 256, 0, stream>>>(A, Bw, bias, C, M, N, K, act);
        }
    };
    auto conv = [&](const float* in, bf16* out, int n) {
        f32_to_bf16_k<<<(n + 255) / 256, 256, 0, stream>>>(in, out, n);
    };

    // ---- weight conversions (bf16 operands for WMMA) ----
    conv(Wpool, Wpool_bf, DE * DE);
    conv(Wproj, Wproj_bf, DD * DE);
    conv(Wq, Wq_bf, DD * DD);   conv(Wk, Wk_bf, DD * DD);   conv(Wv, Wv_bf, DD * DD);
    conv(Wmab, Wmab_bf, DD * DD);
    conv(Wfl, Wfl_bf, DD * 512);
    conv(Win, Win_bf, 3 * DD * DD);

    // ---- 1) embedding masked-mean pool -> bf16 [NN,768] ----
    embed_pool_k<<<NN, 256, 0, stream>>>(input_ids, input_masks, tok_emb, pooled_bf);

    // ---- 2) pooler (tanh) + projection (relu) ----
    gemm(pooled_bf, Wpool_bf, bpool, h1, NN, DE, DE, /*tanh*/2);
    conv(h1, h1_bf, NN * DE);
    gemm(h1_bf, Wproj_bf, bproj, xf, NN, DD, DE, /*relu*/1);
    conv(xf, x_bf, NN * DD);

    // ---- 3) AHT: MAB per group ----
    gemm(x_bf, Wk_bf, bk, Kall, NN, DD, DD, 0);
    gemm(x_bf, Wv_bf, bv, Vall, NN, DD, DD, 0);
    gather_q_k<<<NB * NA, 256, 0, stream>>>(add_ids, xf, qin_bf);
    gemm(qin_bf, Wq_bf, bq, Qf, NB * NA, DD, DD, 0);
    mab_attn_k<<<NB * NH, 128, 0, stream>>>(Qf, Kall, Vall, attn_o);
    ln_k<<<NB * NA, 256, 0, stream>>>(attn_o, nullptr, ln_g1, ln_b1, oln, oln_bf);
    gemm(oln_bf, Wmab_bf, bmab, ffb, NB * NA, DD, DD, /*relu*/1);
    ln_k<<<NB * NA, 256, 0, stream>>>(oln, ffb, ln_g2, ln_b2, xh, nullptr);
    copy_f32_k<<<(NN * DD + 255) / 256, 256, 0, stream>>>(xf, x2, NN * DD);
    scatter_add_k<<<NB * NA, 256, 0, stream>>>(add_ids, xh, x2);

    // ---- 4) FAGCN over 3 graph views ----
    auto fagcn = [&](const int* g, float* outv) {
        const int* src = g; const int* dst = g + NE;
        zero_f32_k<<<(NN + 255) / 256, 256, 0, stream>>>(deg, NN);
        deg_k<<<NE / 256, 256, 0, stream>>>(dst, deg);
        dinv_k<<<(NN + 255) / 256, 256, 0, stream>>>(deg, dinv);
        const float* cur = x2;
        for (int l = 0; l < 2; ++l) {
            const float* Wal = l ? Wal1 : Wal0; const float* balp = l ? bal1 : bal0;
            const float* War = l ? War1 : War0; const float* barp = l ? bar1 : bar0;
            alar_k<<<NN / 8, 256, 0, stream>>>(cur, Wal, balp, War, barp, al, ar);
            zero_f32_k<<<(NN * DD + 255) / 256, 256, 0, stream>>>(agg, NN * DD);
            edge_k<<<NE, 256, 0, stream>>>(src, dst, al, ar, dinv, cur, agg);
            combine_k<<<(NN * DD + 255) / 256, 256, 0, stream>>>(x2, agg, outv, NN * DD);
            cur = outv;
        }
    };
    fagcn(e_g0, fs);
    fagcn(e_g1, fa);
    fagcn(e_g2, fb);

    // ---- 5) gating + concat (bf16) + Wfl (relu) ----
    gating_concat_k<<<NN, 256, 0, stream>>>(x2, Wga, bga, Wgb, bgb, fa, fs, fb, gcat_bf);
    gemm(gcat_bf, Wfl_bf, bfl, xx, NN, DD, 512, /*relu*/1);
    conv(xx, xx_bf, NN * DD);

    // ---- 6) fuse: qkv projection + per-group MHA (row 0) ----
    gemm(xx_bf, Win_bf, bin, qkv, NN, 3 * DD, DD, 0);
    fuse_k<<<NB, 128, 0, stream>>>(qkv, Wout, bout, vec, scores);

    // ---- 7) final MLP head -> pred ----
    head_k<<<NB, 256, 0, stream>>>(vec, W1, b1, W2, b2, pred);
}
